// Layer_Incor_offset_61211873903020
// MI455X (gfx1250) — matrix operand order repaired and checked by reading, not compiled, not run on hardware
//
#include <hip/hip_runtime.h>

// ---------------------------------------------------------------------------
// MI455X (gfx1250) implementation of the deformable-attention layer.
// Dense GEMMs: v_wmma_f32_16x16x32_bf16 (wave32 WMMA, f32 accum) with
// Tensor-Data-Mover (TDM) staging of A/B tiles into padded LDS.
// Softmax / bilinear sampling / layernorm stay f32 VALU.
// Dead q_proj from the reference is skipped.
// ---------------------------------------------------------------------------

typedef __bf16 bf16_t;
typedef __bf16 v16bf __attribute__((ext_vector_type(16)));
typedef float  v8f   __attribute__((ext_vector_type(8)));
typedef unsigned int u32x4 __attribute__((ext_vector_type(4)));
typedef int          i32x8 __attribute__((ext_vector_type(8)));
typedef int          i32x4 __attribute__((ext_vector_type(4)));

#define B_   8
#define HQ_  32
#define WQ_  32
#define D_   512
#define NH_  8
#define KP_  4
#define DK_  64
#define DFF_ 2048
#define M_   (B_*HQ_*WQ_)   // 8192 tokens

#if __has_builtin(__builtin_amdgcn_tensor_load_to_lds) && \
    __has_builtin(__builtin_amdgcn_s_wait_tensorcnt)
#define USE_TDM 1
#else
#define USE_TDM 0
#endif

// ---------------------------- elementwise ----------------------------------
__global__ __launch_bounds__(256)
void k_add_bf16(const float* __restrict__ a, const float* __restrict__ b,
                bf16_t* __restrict__ o, int n) {
  int i = blockIdx.x * 256 + threadIdx.x;
  if (i < n) o[i] = (bf16_t)(a[i] + b[i]);
}

__global__ __launch_bounds__(256)
void k_cvt_bf16(const float* __restrict__ a, bf16_t* __restrict__ o, int n) {
  int i = blockIdx.x * 256 + threadIdx.x;
  if (i < n) o[i] = (bf16_t)a[i];
}

// W: K x N (f32, row major)  ->  Wt: N x K (bf16) so GEMM B-tiles are
// K-contiguous (TDM 2-D tiles with contiguous dim0).
__global__ __launch_bounds__(256)
void k_transpose_bf16(const float* __restrict__ wsrc, bf16_t* __restrict__ wt,
                      int K, int N) {
  int i = blockIdx.x * 256 + threadIdx.x;
  if (i < K * N) {
    int k = i / N, n = i % N;
    wt[(size_t)n * K + k] = (bf16_t)wsrc[i];
  }
}

// ------------------------------ WMMA GEMM ----------------------------------
// C[M x N] = A[M x K] (bf16) * Bt[N x K]^T (bf16) + bias, optional relu,
// optional bf16 duplicate of C.  M fixed = 8192 (multiple of BM).
#define BM 128
#define BN 128
#define BK 32
#define LDT (BK + 8)   // padded LDS row stride: 80 bytes (64 data + 16 pad)

union Frag { uint4 u[2]; v16bf v; };

#if USE_TDM
// Issue a TDM 2-D tile load: tile0 x tile1 bf16 elements, global row stride
// `stride0` (elements), dest LDS rows padded 64B data + 16B pad (matches LDT).
// Rows >= tdim1 read as zero (hardware OOB zero-fill == B-tile N guard).
__device__ __forceinline__ void tdm_load_2d(unsigned lds_addr, const void* gptr,
                                            unsigned tdim0, unsigned tdim1,
                                            unsigned stride0,
                                            unsigned tile0, unsigned tile1)
{
  unsigned long long ga = (unsigned long long)gptr;
  u32x4 g0;
  g0[0] = 1u;                                         // count=1 (valid user D#)
  g0[1] = lds_addr;                                   // LDS byte address
  g0[2] = (unsigned)(ga & 0xffffffffu);               // global_addr[31:0]
  g0[3] = (unsigned)((ga >> 32) & 0x01ffffffu)        // global_addr[56:32]
        | (2u << 30);                                 // type=2 ("image")
  i32x8 g1;
  g1[0] = (int)((1u << 16)      // data_size = 1 -> 2-byte elements
              | (1u << 20)      // pad_enable
              | (3u << 22)      // pad_interval: 16 DWORDs (one 64B row)
              | (3u << 25));    // pad_amount:   4 DWORDs (16B)
  g1[1] = (int)((tdim0 & 0xffffu) << 16);                       // tensor_dim0 lo
  g1[2] = (int)((tdim0 >> 16) | ((tdim1 & 0xffffu) << 16));     // dim0 hi | dim1 lo
  g1[3] = (int)((tdim1 >> 16) | (tile0 << 16));                 // dim1 hi | tile_dim0
  g1[4] = (int)(tile1 & 0xffffu);                               // tile_dim1 (tile_dim2=0)
  g1[5] = (int)stride0;                                         // tensor_dim0_stride lo
  g1[6] = 0;                                                    // stride0 hi | stride1 lo
  g1[7] = 0;
  i32x4 z4 = {0, 0, 0, 0};
  i32x8 z8 = {0, 0, 0, 0, 0, 0, 0, 0};
  // 6-arg form (clang-23 / therock headers): groups 0..4 + cache policy.
  __builtin_amdgcn_tensor_load_to_lds(g0, g1, z4, z4, z8, 0);
}
#endif

__global__ __launch_bounds__(256)
void k_gemm_bf16(const bf16_t* __restrict__ A, const bf16_t* __restrict__ Bt,
                 const float* __restrict__ bias, float* __restrict__ C,
                 bf16_t* __restrict__ Cb, int Ndim, int Kdim, int relu)
{
  __shared__ bf16_t lsA[BM * LDT];
  __shared__ bf16_t lsB[BN * LDT];

  const int tid  = threadIdx.x;
  const int wid  = tid >> 5;       // 8 wave32 per block
  const int lane = tid & 31;
  const int m0   = blockIdx.x * BM;
  const int n0   = blockIdx.y * BN;
  const int l15  = lane & 15;      // A row within 16x32 frag / B column
  const int half = lane >> 4;      // lane half selects K sub-range

  v8f acc[BN / 16];
#pragma unroll
  for (int i = 0; i < BN / 16; ++i) acc[i] = v8f{};

#if USE_TDM
  const unsigned ldsA = (unsigned)(size_t)(void*)lsA;  // LDS aperture addr[31:0]
  const unsigned ldsB = (unsigned)(size_t)(void*)lsB;
#endif

  for (int k0 = 0; k0 < Kdim; k0 += BK) {
#if USE_TDM
    // One wave drives the Tensor Data Mover; TENSORcnt tracks completion.
    if (wid == 0) {
      tdm_load_2d(ldsA, &A[(size_t)m0 * Kdim + k0],
                  (unsigned)(Kdim - k0), (unsigned)(M_ - m0),
                  (unsigned)Kdim, BK, BM);
      tdm_load_2d(ldsB, &Bt[(size_t)n0 * Kdim + k0],
                  (unsigned)(Kdim - k0), (unsigned)(Ndim - n0),
                  (unsigned)Kdim, BK, BN);
      __builtin_amdgcn_s_wait_tensorcnt(0);
    }
#else
    // Fallback: cooperative 128-bit staging (round-1 path).
#pragma unroll
    for (int i = 0; i < 2; ++i) {
      int idx = tid + i * 256;
      int row = idx >> 2;
      int c8  = (idx & 3) << 3;
      *reinterpret_cast<uint4*>(&lsA[row * LDT + c8]) =
          *reinterpret_cast<const uint4*>(&A[(size_t)(m0 + row) * Kdim + k0 + c8]);
    }
#pragma unroll
    for (int i = 0; i < 2; ++i) {
      int idx = tid + i * 256;
      int row = idx >> 2;
      int c8  = (idx & 3) << 3;
      uint4 v = {0u, 0u, 0u, 0u};
      int n = n0 + row;
      if (n < Ndim)
        v = *reinterpret_cast<const uint4*>(&Bt[(size_t)n * Kdim + k0 + c8]);
      *reinterpret_cast<uint4*>(&lsB[row * LDT + c8]) = v;
    }
#endif
    __syncthreads();

    // A fragment (16x32 bf16): lane half h row m holds K = h*8+0..7 and
    // 16+h*8+0..7  -> two contiguous 16B LDS loads.
    Frag af;
    {
      const bf16_t* pa = &lsA[(wid * 16 + l15) * LDT + half * 8];
      af.u[0] = *reinterpret_cast<const uint4*>(pa);
      af.u[1] = *reinterpret_cast<const uint4*>(pa + 16);
    }
    // Preload all 8 B fragments, then burst the 8 WMMAs back-to-back.
    Frag bfr[BN / 16];
#pragma unroll
    for (int nt = 0; nt < BN / 16; ++nt) {
      const bf16_t* pb = &lsB[(nt * 16 + l15) * LDT + half * 8];
      bfr[nt].u[0] = *reinterpret_cast<const uint4*>(pb);
      bfr[nt].u[1] = *reinterpret_cast<const uint4*>(pb + 16);
    }
#pragma unroll
    for (int nt = 0; nt < BN / 16; ++nt) {
      acc[nt] = __builtin_amdgcn_wmma_f32_16x16x32_bf16(
          false, af.v, false, bfr[nt].v, (short)0, acc[nt], false, false);
    }
    __syncthreads();
  }

  // Epilogue: C/D layout => lane n = lane&15, M = r + 8*(lane>>4).
#pragma unroll
  for (int nt = 0; nt < BN / 16; ++nt) {
    int n = n0 + nt * 16 + l15;
    if (n < Ndim) {
      float bv = bias[n];
#pragma unroll
      for (int r = 0; r < 8; ++r) {
        int m = m0 + wid * 16 + half * 8 + r;
        float v = acc[nt][r] + bv;
        if (relu) v = v > 0.f ? v : 0.f;
        C[(size_t)m * Ndim + n] = v;
        if (Cb) Cb[(size_t)m * Ndim + n] = (bf16_t)v;
      }
    }
  }
}

// ------------------------------ softmax (len 4) ----------------------------
__global__ __launch_bounds__(256)
void k_softmax4(float* __restrict__ p, int rows) {
  int r = blockIdx.x * 256 + threadIdx.x;
  if (r < rows) {
    float4 v = *reinterpret_cast<float4*>(&p[(size_t)r * 4]);
    float mx = fmaxf(fmaxf(v.x, v.y), fmaxf(v.z, v.w));
    float ex = __expf(v.x - mx), ey = __expf(v.y - mx);
    float ez = __expf(v.z - mx), ew = __expf(v.w - mx);
    float inv = 1.f / (ex + ey + ez + ew);
    float4 o = {ex * inv, ey * inv, ez * inv, ew * inv};
    *reinterpret_cast<float4*>(&p[(size_t)r * 4]) = o;
  }
}

// --------------------- deformable bilinear sampling ------------------------
// Each thread owns one (b,h,y,x) query and 2 consecutive channels.
// Replicates the reference exactly, including its ref-tile indexing
// (ref_rep[n] = ref[(b*NH+h) % B]) and align_corners=False unnormalization.
__global__ __launch_bounds__(256)
void k_deform_sample(const float* __restrict__ kv,    // (B,HQ,WQ, NH*DK)
                     const float* __restrict__ offs,  // (M, NH*KP*2)
                     const float* __restrict__ Aw,    // (M, NH*KP) softmaxed
                     const float* __restrict__ ref,   // (B,HQ,WQ,2)
                     float* __restrict__ feat,        // (M, D)
                     bf16_t* __restrict__ featb)
{
  int gid = blockIdx.x * 256 + threadIdx.x;  // B*NH*HQ*WQ*32 threads
  int dk  = (gid & 31) * 2;
  int q   = gid >> 5;                        // ((b*NH+h)*HQ + y)*WQ + x
  int x = q & 31;
  int y = (q >> 5) & 31;
  int h = (q >> 10) & 7;
  int b = q >> 13;
  int token = (b * HQ_ + y) * WQ_ + x;
  int rb = (b * NH_ + h) % B_;               // reference's tile() batch pick
  const float* rp = &ref[((size_t)(rb * HQ_ + y) * WQ_ + x) * 2];
  float px = rp[0] * (WQ_ - 1);
  float py = rp[1] * (HQ_ - 1);
  const float* op = &offs[(size_t)token * (2 * NH_ * KP_) + h * KP_ * 2];
  const float* ap = &Aw[(size_t)token * (NH_ * KP_) + h * KP_];
  const float* fm = &kv[(size_t)b * HQ_ * WQ_ * D_ + h * DK_ + dk];

  float f0 = 0.f, f1 = 0.f;
#pragma unroll
  for (int k = 0; k < KP_; ++k) {
    float ptx = px + op[k * 2 + 0];
    float pty = py + op[k * 2 + 1];
    float vx = 2.f * ptx / (WQ_ - 1) - 1.f;
    float vy = 2.f * pty / (HQ_ - 1) - 1.f;
    float ix = ((vx + 1.f) * WQ_ - 1.f) * 0.5f;   // align_corners=False
    float iy = ((vy + 1.f) * HQ_ - 1.f) * 0.5f;
    float x0f = floorf(ix), y0f = floorf(iy);
    float wx1 = ix - x0f, wy1 = iy - y0f;
    float wx0 = 1.f - wx1, wy0 = 1.f - wy1;
    float a = ap[k];
    auto corner = [&](float xf, float yf, float wgt) {
      bool valid = (xf >= 0.f) && (xf <= (float)(WQ_ - 1)) &&
                   (yf >= 0.f) && (yf <= (float)(HQ_ - 1));
      int xi = (int)fminf(fmaxf(xf, 0.f), (float)(WQ_ - 1));
      int yi = (int)fminf(fmaxf(yf, 0.f), (float)(HQ_ - 1));
      float w2 = valid ? (a * wgt) : 0.f;
      const float* pv = fm + (size_t)(yi * WQ_ + xi) * D_;
      f0 += w2 * pv[0];
      f1 += w2 * pv[1];
    };
    corner(x0f,       y0f,       wx0 * wy0);
    corner(x0f + 1.f, y0f,       wx1 * wy0);
    corner(x0f,       y0f + 1.f, wx0 * wy1);
    corner(x0f + 1.f, y0f + 1.f, wx1 * wy1);
  }
  size_t fo = (size_t)token * D_ + h * DK_ + dk;  // (B,HQ,WQ,NH,DK) == (M,D)
  feat[fo]     = f0;
  feat[fo + 1] = f1;
  featb[fo]     = (bf16_t)f0;
  featb[fo + 1] = (bf16_t)f1;
}

// ------------------------- residual + LayerNorm ----------------------------
__global__ __launch_bounds__(256)
void k_add_layernorm(const float* __restrict__ a, const float* __restrict__ bsrc,
                     const float* __restrict__ g, const float* __restrict__ beta,
                     float* __restrict__ y)
{
  __shared__ float s_sum[8], s_sq[8];
  int t = threadIdx.x;
  size_t base = (size_t)blockIdx.x * D_;
  float v0 = a[base + t]       + bsrc[base + t];
  float v1 = a[base + t + 256] + bsrc[base + t + 256];
  float s = v0 + v1, s2 = v0 * v0 + v1 * v1;
  for (int o = 16; o > 0; o >>= 1) {           // wave32 tree reduce
    s  += __shfl_down(s, o);
    s2 += __shfl_down(s2, o);
  }
  int lane = t & 31, wid = t >> 5;
  if (lane == 0) { s_sum[wid] = s; s_sq[wid] = s2; }
  __syncthreads();
  if (t == 0) {
    float ts = 0.f, ts2 = 0.f;
    for (int i = 0; i < 8; ++i) { ts += s_sum[i]; ts2 += s_sq[i]; }
    s_sum[0] = ts; s_sq[0] = ts2;
  }
  __syncthreads();
  float mean = s_sum[0] / (float)D_;
  float var  = s_sq[0] / (float)D_ - mean * mean;
  float inv  = rsqrtf(var + 1e-5f);
  y[base + t]       = (v0 - mean) * inv * g[t]       + beta[t];
  y[base + t + 256] = (v1 - mean) * inv * g[t + 256] + beta[t + 256];
}

// ------------------------------- launcher ----------------------------------
extern "C" void kernel_launch(void* const* d_in, const int* in_sizes, int n_in,
                              void* d_out, int out_size, void* d_ws, size_t ws_size,
                              hipStream_t stream)
{
  const float* src   = (const float*)d_in[0];
  const float* ref   = (const float*)d_in[1];
  const float* pos   = (const float*)d_in[2];
  const float* srcq  = (const float*)d_in[3];
  // d_in[4], d_in[5] = q_w / q_b : dead code in reference, skipped.
  const float* b_w   = (const float*)d_in[6];
  const float* b_b   = (const float*)d_in[7];
  const float* k_w   = (const float*)d_in[8];
  const float* k_b   = (const float*)d_in[9];
  const float* off_w = (const float*)d_in[10];
  const float* off_b = (const float*)d_in[11];
  const float* A_w   = (const float*)d_in[12];
  const float* A_b   = (const float*)d_in[13];
  const float* wm_w  = (const float*)d_in[14];
  const float* wm_b  = (const float*)d_in[15];
  const float* fc1_w = (const float*)d_in[16];
  const float* fc1_b = (const float*)d_in[17];
  const float* fc2_w = (const float*)d_in[18];
  const float* fc2_b = (const float*)d_in[19];
  const float* ln_g  = (const float*)d_in[20];
  const float* ln_b  = (const float*)d_in[21];
  float* outy = (float*)d_out;

  char* wsp = (char*)d_ws;
  auto take = [&](size_t bytes) {
    void* p = (void*)wsp;
    wsp += (bytes + 255) & ~(size_t)255;
    return p;
  };

  bf16_t* xb    = (bf16_t*)take((size_t)M_ * D_ * 2);
  bf16_t* sqb   = (bf16_t*)take((size_t)M_ * D_ * 2);
  bf16_t* bqb   = (bf16_t*)take((size_t)M_ * D_ * 2);
  bf16_t* featb = (bf16_t*)take((size_t)M_ * D_ * 2);
  bf16_t* outb  = (bf16_t*)take((size_t)M_ * D_ * 2);
  bf16_t* ff1b  = (bf16_t*)take((size_t)M_ * DFF_ * 2);
  bf16_t* WbT   = (bf16_t*)take((size_t)D_ * D_ * 2);
  bf16_t* WkT   = (bf16_t*)take((size_t)D_ * D_ * 2);
  bf16_t* WmT   = (bf16_t*)take((size_t)D_ * D_ * 2);
  bf16_t* Wf1T  = (bf16_t*)take((size_t)DFF_ * D_ * 2);
  bf16_t* Wf2T  = (bf16_t*)take((size_t)D_ * DFF_ * 2);
  bf16_t* WoffT = (bf16_t*)take((size_t)64 * D_ * 2);
  bf16_t* WAT   = (bf16_t*)take((size_t)32 * D_ * 2);
  float* bq   = (float*)take((size_t)M_ * D_ * 4);
  float* kvb  = (float*)take((size_t)M_ * D_ * 4);
  float* offv = (float*)take((size_t)M_ * 64 * 4);
  float* Awv  = (float*)take((size_t)M_ * 32 * 4);
  float* feat = (float*)take((size_t)M_ * D_ * 4);
  float* outp = (float*)take((size_t)M_ * D_ * 4);
  float* ff1  = (float*)take((size_t)M_ * DFF_ * 4);
  float* ff2  = (float*)take((size_t)M_ * D_ * 4);

  const int n_md = M_ * D_;
  k_add_bf16<<<n_md / 256, 256, 0, stream>>>(src, pos, xb, n_md);
  k_cvt_bf16<<<n_md / 256, 256, 0, stream>>>(srcq, sqb, n_md);

  k_transpose_bf16<<<(D_ * D_ + 255) / 256, 256, 0, stream>>>(b_w,  WbT,  D_, D_);
  k_transpose_bf16<<<(D_ * D_ + 255) / 256, 256, 0, stream>>>(k_w,  WkT,  D_, D_);
  k_transpose_bf16<<<(D_ * D_ + 255) / 256, 256, 0, stream>>>(wm_w, WmT,  D_, D_);
  k_transpose_bf16<<<(D_ * DFF_ + 255) / 256, 256, 0, stream>>>(fc1_w, Wf1T, D_, DFF_);
  k_transpose_bf16<<<(DFF_ * D_ + 255) / 256, 256, 0, stream>>>(fc2_w, Wf2T, DFF_, D_);
  k_transpose_bf16<<<(D_ * 64 + 255) / 256, 256, 0, stream>>>(off_w, WoffT, D_, 64);
  k_transpose_bf16<<<(D_ * 32 + 255) / 256, 256, 0, stream>>>(A_w,  WAT,  D_, 32);

  dim3 blk(256);
  dim3 g512(M_ / BM, (D_  + BN - 1) / BN);   // 64 x 4
  dim3 gsm (M_ / BM, 1);                     // small-N GEMMs
  dim3 gff (M_ / BM, (DFF_ + BN - 1) / BN);  // 64 x 16

  // bq = src_query @ b_w + b_b   (also bf16 copy for downstream GEMMs)
  k_gemm_bf16<<<g512, blk, 0, stream>>>(sqb, WbT, b_b, bq, bqb, D_, D_, 0);
  // off = bq @ off_w + off_b  (N=64; TDM OOB zero-fill handles the tail)
  k_gemm_bf16<<<gsm, blk, 0, stream>>>(bqb, WoffT, off_b, offv, nullptr, 64, D_, 0);
  // A logits = bq @ A_w + A_b (N=32), then softmax over S*KP=4
  k_gemm_bf16<<<gsm, blk, 0, stream>>>(bqb, WAT, A_b, Awv, nullptr, 32, D_, 0);
  k_softmax4<<<(M_ * NH_ + 255) / 256, 256, 0, stream>>>(Awv, M_ * NH_);
  // kv = (src+pos) @ k_w + k_b
  k_gemm_bf16<<<g512, blk, 0, stream>>>(xb, WkT, k_b, kvb, nullptr, D_, D_, 0);
  // deformable bilinear sampling -> feat (B,HQ,WQ,D)
  k_deform_sample<<<(B_ * NH_ * HQ_ * WQ_ * 32) / 256, 256, 0, stream>>>(
      kvb, offv, Awv, ref, feat, featb);
  // out = feat @ wm_w + wm_b
  k_gemm_bf16<<<g512, blk, 0, stream>>>(featb, WmT, wm_b, outp, outb, D_, D_, 0);
  // ff1 = relu(out @ fc1_w + fc1_b)
  k_gemm_bf16<<<gff, blk, 0, stream>>>(outb, Wf1T, fc1_b, ff1, ff1b, DFF_, D_, 1);
  // ff2 = ff1 @ fc2_w + fc2_b
  k_gemm_bf16<<<g512, blk, 0, stream>>>(ff1b, Wf2T, fc2_b, ff2, nullptr, D_, DFF_, 0);
  // y = LN(out + ff2) * g + b
  k_add_layernorm<<<M_, 256, 0, stream>>>(outp, ff2, ln_g, ln_b, outy);
}
